// DeformableTransformer_72559177499131
// MI455X (gfx1250) — matrix operand order repaired and checked by reading, not compiled, not run on hardware
//
#include <hip/hip_runtime.h>
#include <hip/hip_bf16.h>
#include <math.h>

#define LQq    4000
#define DMM    256
#define NHH    8
#define NLL    4
#define NPP    4
#define DHH    32
#define LEN_IN 37440
#define KDIM   256

typedef __bf16 bf16;
typedef __attribute__((ext_vector_type(16))) __bf16 v16bf;
typedef __attribute__((ext_vector_type(8)))  __bf16 v8bf;
typedef __attribute__((ext_vector_type(8)))  float  v8f;

// ---------- f32 -> bf16 (round to nearest even) ----------
__device__ __forceinline__ bf16 f2bf(float f) {
  unsigned u = __builtin_bit_cast(unsigned, f);
  unsigned r = u + 0x7FFFu + ((u >> 16) & 1u);
  unsigned short h = (unsigned short)(r >> 16);
  return __builtin_bit_cast(bf16, h);
}

__global__ void cvt_f32_bf16(const float* __restrict__ src, bf16* __restrict__ dst, long n) {
  long i = (long)blockIdx.x * blockDim.x + threadIdx.x;
  if (i < n) dst[i] = f2bf(src[i]);
}

// src is [K][N] row-major f32; dst is [N][K] row-major bf16 (Bt for WMMA B-fragments)
__global__ void transpose_cvt(const float* __restrict__ src, bf16* __restrict__ dst, int K, int N) {
  int i = blockIdx.x * blockDim.x + threadIdx.x;
  if (i >= K * N) return;
  int k = i / N, n = i - k * N;
  dst[(size_t)n * K + k] = f2bf(src[i]);
}

union Frag { v16bf v; struct { v8bf lo, hi; } h; };

__device__ __forceinline__ void loadA(Frag& f, const bf16* arow, int k0, int half) {
  f.h.lo = *(const v8bf*)(arow + k0 + half * 8);
  f.h.hi = *(const v8bf*)(arow + k0 + 16 + half * 8);
}

// ---------- bf16 WMMA GEMM:  C[M][N] = A[M][KC] * Bt[N][KC]^T + bias ----------
// Block = 256 threads = 8 waves; block tile = 128(M) x 64(N); KC compile-time.
// The 64x KC bf16 weight strip shared by all 8 waves is staged once into LDS
// (row-padded 16B -> conflict-free ds_load_b128 fragments), so B fragments
// come from ~30-cycle LDS (DScnt) while streamed A fragments prefetch through
// global (LOADcnt) -- the WMMAs never drain the global pipe for weights.
// Requires: M % 16 == 0, N % 64 == 0, KC % 32 == 0.
template<int KC>
__global__ __launch_bounds__(256)
void gemm_bf16_wmma(const bf16* __restrict__ A, const bf16* __restrict__ Bt,
                    const float* __restrict__ bias, float* __restrict__ C,
                    int M, int N)
{
  constexpr int ROWPAD = 8;            // 8 bf16 = 16B pad per row
  constexpr int LROW = KC + ROWPAD;    // 264 elems = 528B: lanes -> distinct banks
  __shared__ __align__(16) bf16 smem[64 * LROW];

  const int tid  = threadIdx.x;
  const int wave = tid >> 5;
  const int lane = tid & 31;
  const int half = lane >> 4;   // 0: lanes 0-15, 1: lanes 16-31
  const int l16  = lane & 15;
  const int tileM = blockIdx.y * 8 + wave;
  const int n0 = blockIdx.x * 64;

  // ---- cooperative stage: Bt rows n0..n0+63 -> LDS (32KB, once per block) ----
  {
    const int r  = tid >> 2;                 // 0..63
    const int cq = (tid & 3) * (KC / 4);     // 0, 64, 128, 192
    const bf16* g = Bt + (size_t)(n0 + r) * KC + cq;
    bf16* s = smem + r * LROW + cq;
#pragma unroll
    for (int i = 0; i < KC / 4; i += 8)
      *(v8bf*)(s + i) = *(const v8bf*)(g + i);
  }
  __syncthreads();

  if (tileM * 16 >= M) return;

  const bf16* arow  = A + (size_t)(tileM * 16 + l16) * KC;
  const bf16* brow0 = smem + l16 * LROW + half * 8;    // t stride = 16*LROW elems

  v8f acc[4] = {};

  // 2-deep pipelined A fragments (global); B fragments read from LDS per step
  Frag a[2];
  loadA(a[0], arow, 0, half);

  constexpr int STEPS = KC / 32;
#pragma unroll
  for (int s = 0; s < STEPS; ++s) {
    const int cur = s & 1;
    const int nxt = cur ^ 1;
    if (s + 1 < STEPS) loadA(a[nxt], arow, (s + 1) * 32, half);
    const int k0 = s * 32;
#pragma unroll
    for (int t = 0; t < 4; ++t) {
      const bf16* p = brow0 + t * (16 * LROW) + k0;
      Frag b;
      b.h.lo = *(const v8bf*)(p);
      b.h.hi = *(const v8bf*)(p + 16);
      acc[t] = __builtin_amdgcn_wmma_f32_16x16x32_bf16(
          /*neg_a=*/false, a[cur].v, /*neg_b=*/false, b.v,
          /*c_mod=*/(short)0, acc[t], /*reuse_a=*/false, /*reuse_b=*/false);
    }
  }

  // C/D layout: VGPR r, lane -> C[r + 8*half][l16]
#pragma unroll
  for (int t = 0; t < 4; ++t) {
    const int col = n0 + t * 16 + l16;
    const float bb = bias[col];
#pragma unroll
    for (int r = 0; r < 8; ++r) {
      C[(size_t)(tileM * 16 + r + 8 * half) * N + col] = acc[t][r] + bb;
    }
  }
}

// ---------- softmax over groups of 16 (per query,head) in-place ----------
__global__ void softmax16(float* __restrict__ a, int rows) {
  int r = blockIdx.x * blockDim.x + threadIdx.x;
  if (r >= rows) return;
  float* p = a + (size_t)r * 16;
  float m = p[0];
#pragma unroll
  for (int i = 1; i < 16; ++i) m = fmaxf(m, p[i]);
  float e[16]; float s = 0.f;
#pragma unroll
  for (int i = 0; i < 16; ++i) { e[i] = __expf(p[i] - m); s += e[i]; }
  float inv = 1.f / s;
#pragma unroll
  for (int i = 0; i < 16; ++i) p[i] = e[i] * inv;
}

// ---------- trilinear deformable sampling ----------
// value [LEN_IN][NH*DH] f32; ref [LQ][NL][3]; off [LQ][NH][NL][NP][3];
// attn [LQ][NH][NL][NP]; out [LQ][NH*DH]
__global__ __launch_bounds__(256)
void sample_kernel(const float* __restrict__ value, const float* __restrict__ ref,
                   const float* __restrict__ off, const float* __restrict__ attn,
                   float* __restrict__ out)
{
  const int dims[4] = {32, 16, 8, 4};
  const int q = blockIdx.x;
  const int h = threadIdx.y;   // 0..7
  const int c = threadIdx.x;   // 0..31 (channel lane -> coalesced gathers)
  float acc = 0.f;
  int start = 0;
#pragma unroll
  for (int l = 0; l < NLL; ++l) {
    const int S = dims[l];
    const int HW = S * S;
    const float rx = ref[((size_t)q * NLL + l) * 3 + 0];
    const float ry = ref[((size_t)q * NLL + l) * 3 + 1];
    const float rz = ref[((size_t)q * NLL + l) * 3 + 2];
    const float invS = 1.0f / (float)S;  // normalizer (D,W,H); cubic levels => all equal
#pragma unroll
    for (int p = 0; p < NPP; ++p) {
      const size_t ob = (size_t)q * (NHH * NLL * NPP * 3) + h * (NLL * NPP * 3) + l * (NPP * 3) + p * 3;
      const float lx = rx + off[ob + 0] * invS;
      const float ly = ry + off[ob + 1] * invS;
      const float lz = rz + off[ob + 2] * invS;
      const float aw = attn[(size_t)q * (NHH * NLL * NPP) + h * (NLL * NPP) + l * NPP + p];
      const float x = lx * S - 0.5f, y = ly * S - 0.5f, z = lz * S - 0.5f;
      const float x0f = floorf(x), y0f = floorf(y), z0f = floorf(z);
      const float fx = x - x0f, fy = y - y0f, fz = z - z0f;
      const int x0 = (int)x0f, y0 = (int)y0f, z0 = (int)z0f;
#pragma unroll
      for (int dz = 0; dz < 2; ++dz) {
        const int zi = z0 + dz;
        if (zi < 0 || zi >= S) continue;
        const float wz = dz ? fz : 1.f - fz;
#pragma unroll
        for (int dy = 0; dy < 2; ++dy) {
          const int yi = y0 + dy;
          if (yi < 0 || yi >= S) continue;
          const float wy = dy ? fy : 1.f - fy;
#pragma unroll
          for (int dx = 0; dx < 2; ++dx) {
            const int xi = x0 + dx;
            if (xi < 0 || xi >= S) continue;
            const float wx = dx ? fx : 1.f - fx;
            const int idx = zi * HW + yi * S + xi;
            acc += wx * wy * wz * aw * value[(size_t)(start + idx) * DMM + h * DHH + c];
          }
        }
      }
    }
    start += S * S * S;
  }
  out[(size_t)q * DMM + h * DHH + c] = acc;
}

// ---------------------------------------------------------------------------
extern "C" void kernel_launch(void* const* d_in, const int* in_sizes, int n_in,
                              void* d_out, int out_size, void* d_ws, size_t ws_size,
                              hipStream_t stream)
{
  const float* query   = (const float*)d_in[0];
  const float* refpts  = (const float*)d_in[1];
  const float* inflat  = (const float*)d_in[2];
  const float* w_value = (const float*)d_in[3];
  const float* b_value = (const float*)d_in[4];
  const float* w_off   = (const float*)d_in[5];
  const float* b_off   = (const float*)d_in[6];
  const float* w_attn  = (const float*)d_in[7];
  const float* b_attn  = (const float*)d_in[8];
  const float* w_out   = (const float*)d_in[9];
  const float* b_out   = (const float*)d_in[10];
  float* out = (float*)d_out;

  unsigned char* ws = (unsigned char*)d_ws;
  size_t cur = 0;
  auto alloc = [&](size_t bytes) -> void* {
    void* p = ws + cur;
    cur += (bytes + 255) & ~(size_t)255;
    return p;
  };
  bf16*  A_v   = (bf16*)alloc((size_t)LEN_IN * DMM * 2);
  bf16*  A_q   = (bf16*)alloc((size_t)LQq * DMM * 2);
  bf16*  Wv_t  = (bf16*)alloc((size_t)DMM * DMM * 2);
  bf16*  Wo_t  = (bf16*)alloc((size_t)DMM * 384 * 2);
  bf16*  Wa_t  = (bf16*)alloc((size_t)DMM * 128 * 2);
  bf16*  Wp_t  = (bf16*)alloc((size_t)DMM * DMM * 2);
  float* value = (float*)alloc((size_t)LEN_IN * DMM * 4);
  float* offb  = (float*)alloc((size_t)LQq * 384 * 4);
  float* attnb = (float*)alloc((size_t)LQq * 128 * 4);
  float* samp  = (float*)alloc((size_t)LQq * DMM * 4);
  bf16*  sampb = (bf16*)alloc((size_t)LQq * DMM * 2);

  // 1) convert activations to bf16
  { long n = (long)LEN_IN * DMM; cvt_f32_bf16<<<(unsigned)((n + 255) / 256), 256, 0, stream>>>(inflat, A_v, n); }
  { long n = (long)LQq * DMM;    cvt_f32_bf16<<<(unsigned)((n + 255) / 256), 256, 0, stream>>>(query, A_q, n); }

  // 2) transpose+convert weights to bf16 [N][K]
  transpose_cvt<<<(DMM * DMM + 255) / 256, 256, 0, stream>>>(w_value, Wv_t, DMM, DMM);
  transpose_cvt<<<(DMM * 384 + 255) / 256, 256, 0, stream>>>(w_off,  Wo_t, DMM, 384);
  transpose_cvt<<<(DMM * 128 + 255) / 256, 256, 0, stream>>>(w_attn, Wa_t, DMM, 128);
  transpose_cvt<<<(DMM * DMM + 255) / 256, 256, 0, stream>>>(w_out,  Wp_t, DMM, DMM);

  auto gemm = [&](const bf16* A, const bf16* Bt, const float* bias, float* C, int M, int N) {
    dim3 grid(N / 64, (unsigned)(((M / 16) + 7) / 8));
    gemm_bf16_wmma<KDIM><<<grid, 256, 0, stream>>>(A, Bt, bias, C, M, N);
  };

  // 3) projections
  gemm(A_v, Wv_t, b_value, value, LEN_IN, DMM);   // value projection (dominant GEMM)
  gemm(A_q, Wo_t, b_off,  offb,  LQq, 384);       // sampling offsets
  gemm(A_q, Wa_t, b_attn, attnb, LQq, 128);       // attention logits

  // 4) softmax over NL*NP=16 per (q,h)
  softmax16<<<(LQq * NHH + 255) / 256, 256, 0, stream>>>(attnb, LQq * NHH);

  // 5) deformable trilinear sampling
  dim3 sblk(32, 8);
  sample_kernel<<<LQq, sblk, 0, stream>>>(value, refpts, offb, attnb, samp);

  // 6) output projection
  { long n = (long)LQq * DMM; cvt_f32_bf16<<<(unsigned)((n + 255) / 256), 256, 0, stream>>>(samp, sampb, n); }
  gemm(sampb, Wp_t, b_out, out, LQq, DMM);

  (void)in_sizes; (void)n_in; (void)out_size; (void)ws_size;
}
